// STEncoder_54906861912567
// MI455X (gfx1250) — matrix operand order repaired and checked by reading, not compiled, not run on hardware
//
#include <hip/hip_runtime.h>
#include <hip/hip_bf16.h>
#include <math.h>

// ---------------------------------------------------------------------------
// Set-Transformer encoder for MI455X (gfx1250), wave32, WMMA f16->f32.
// B=4, N=2048, D=128, H=4 (dh=32), D_OUT=256.
// ---------------------------------------------------------------------------

typedef _Float16 v16h __attribute__((ext_vector_type(16)));
typedef _Float16 v8h  __attribute__((ext_vector_type(8)));
typedef _Float16 v4h  __attribute__((ext_vector_type(4)));
typedef float    v8f  __attribute__((ext_vector_type(8)));
typedef int      v4i  __attribute__((ext_vector_type(4)));

// Pointer types for the async global->LDS builtin (signature revealed by the
// compiler: param0 = v4i in global AS, non-const).
typedef __attribute__((address_space(1))) v4i* as1_v4i_ptr;
typedef __attribute__((address_space(3))) v4i* as3_v4i_ptr;

#if __has_builtin(__builtin_amdgcn_global_load_async_to_lds_b128)
#define HAVE_ASYNC_LDS 1
#else
#define HAVE_ASYNC_LDS 0
#endif

static __device__ inline v8f wmma_f16(v16h a, v16h b, v8f c) {
  // D = A(16x32 f16) * B(32x16 f16) + C(16x16 f32)
  return __builtin_amdgcn_wmma_f32_16x16x32_f16(
      /*neg_a=*/false, a, /*neg_b=*/false, b,
      /*c_mod=*/(short)0, c, /*reuse_a=*/false, /*reuse_b=*/false);
}

static __device__ inline v16h ld_a_pair(const _Float16* p0, const _Float16* p1) {
  v16h r;
  reinterpret_cast<v8h*>(&r)[0] = *reinterpret_cast<const v8h*>(p0);
  reinterpret_cast<v8h*>(&r)[1] = *reinterpret_cast<const v8h*>(p1);
  return r;
}

static __device__ inline void wait_asynccnt0() {
#if HAVE_ASYNC_LDS
#if __has_builtin(__builtin_amdgcn_s_wait_asynccnt)
  __builtin_amdgcn_s_wait_asynccnt(0);
#else
  asm volatile("s_wait_asynccnt 0x0" ::: "memory");
#endif
#endif
}

// -------------------------- conversion kernels ------------------------------

__global__ void k_cvt(const float* __restrict__ src, _Float16* __restrict__ dst, int n) {
  int i = blockIdx.x * blockDim.x + threadIdx.x;
  if (i < n) dst[i] = (_Float16)src[i];
}

// src [Kd, Nc] row-major -> dst [Nc, Kd] (transposed) f16
__global__ void k_cvtT(const float* __restrict__ src, _Float16* __restrict__ dst,
                       int Kd, int Nc) {
  int i = blockIdx.x * blockDim.x + threadIdx.x;
  if (i < Kd * Nc) {
    int k = i / Nc, n = i % Nc;
    dst[n * Kd + k] = (_Float16)src[i];
  }
}

// ------------------------------ WMMA GEMM -----------------------------------
// out[M,Nc] (f16) = act( A[M,Kd](f16) @ Wt^T + bias ) [+ res]
// Wt stored transposed: [Nc, Kd].
// One 16x64 output strip per wave (4 accumulators, A-tile reused 4x per K-step).
__global__ __launch_bounds__(256)
void k_gemm(const _Float16* __restrict__ A, const _Float16* __restrict__ Wt,
            const float* __restrict__ bias, _Float16* __restrict__ out,
            const _Float16* __restrict__ res, int M, int Kd, int Nc, int relu) {
  const int wave = threadIdx.x >> 5;
  const int lane = threadIdx.x & 31;
  const int lr = lane & 15;      // row (A) / column (B,D)
  const int lh = lane >> 4;      // lane-half selector
  const int nsup = Nc >> 6;      // 64-column super-tiles
  const int tile = blockIdx.x * (blockDim.x >> 5) + wave;
  const int tm = tile / nsup;
  const int ts = tile % nsup;
  if (tm * 16 >= M) return;      // wave-uniform

  const long row0 = (long)tm * 16;
  const int  n0   = ts * 64;
  const _Float16* arow = A + (row0 + lr) * (long)Kd;

  v8f acc[4] = {{}, {}, {}, {}};
  for (int k0 = 0; k0 < Kd; k0 += 32) {
    // A 16x32: lane<16 holds K {k0..k0+7, k0+16..k0+23}; lane>=16 the other halves
    v16h a = ld_a_pair(arow + k0 + lh * 8, arow + k0 + 16 + lh * 8);
#pragma unroll
    for (int nt = 0; nt < 4; nt++) {
      // B 32x16: column = lane%16, K = {lh*8..+7, 16+lh*8..+7} (same order as A)
      const _Float16* wrow = Wt + (long)(n0 + nt * 16 + lr) * Kd + k0 + lh * 8;
      v16h b = ld_a_pair(wrow, wrow + 16);
      acc[nt] = wmma_f16(a, b, acc[nt]);
    }
  }

  // D layout: element v -> row = row0 + v + 8*(lane/16), col = n0 + nt*16 + lane%16
#pragma unroll
  for (int nt = 0; nt < 4; nt++) {
    const int col = n0 + nt * 16 + lr;
    const float bcol = bias[col];
#pragma unroll
    for (int v = 0; v < 8; v++) {
      long row = row0 + v + 8 * lh;
      float y = acc[nt][v] + bcol;
      if (relu) y = fmaxf(y, 0.0f);
      if (res)  y += (float)res[row * Nc + col];
      out[row * Nc + col] = (_Float16)y;
    }
  }
}

// --------------------------- fused sigmoid attention ------------------------
// O = Qh + sigmoid(Qh Kh^T / sqrt(D)) Vh  per (batch, head).
// One block = one (b, h, 128-query chunk); 8 waves x 16 query rows.
// K 32-key tile staged via async global->LDS copy (ASYNCcnt); V staged
// transposed through registers (scatter).
__global__ __launch_bounds__(256)
void k_attn(const _Float16* __restrict__ Q, const _Float16* __restrict__ K,
            const _Float16* __restrict__ V, _Float16* __restrict__ O,
            int Bn, int Nn, int Dn) {
  constexpr int Hh = 4, dh = 32;
  __shared__ __align__(16) _Float16 kt[32 * 32];        // [key][dim]
  __shared__ __align__(16) _Float16 vt[32 * 32];        // [dim][key] (transposed)
  __shared__ __align__(16) _Float16 sc[8][16 * 32];     // per-wave score scratch

  const int qchunks = Nn / 128;
  const int unit = blockIdx.x;
  const int b  = unit / (Hh * qchunks);
  const int h  = (unit / qchunks) % Hh;
  const int qc = unit % qchunks;
  const int wave = threadIdx.x >> 5;
  const int lane = threadIdx.x & 31;
  const int lr = lane & 15, lh = lane >> 4;

  const long rowbase = (long)b * Nn + qc * 128 + wave * 16;
  const float scale = 0.08838834764831845f;             // 1/sqrt(128)

  // Load Q A-tile once (16 queries x 32 head-dims)
  const _Float16* qrow = Q + (rowbase + lr) * (long)Dn + h * dh;
  v16h qa = ld_a_pair(qrow + lh * 8, qrow + 16 + lh * 8);

  v8f acc0 = {}, acc1 = {};                             // O dims 0-15 / 16-31

  for (int kb = 0; kb < Nn; kb += 32) {
    // ---- cooperative staging of 32 keys ----
    {
      const int t = threadIdx.x;
      const int key = t >> 3;
      const int c4  = (t & 7) * 4;
      const long src = ((long)b * Nn + kb + key) * Dn + h * dh + c4;
      // V: transpose through registers (scatter stores)
      v4h vv = *reinterpret_cast<const v4h*>(V + src);
#pragma unroll
      for (int j = 0; j < 4; j++) vt[(c4 + j) * 32 + key] = vv[j];
#if HAVE_ASYNC_LDS
      // K: row-major copy, async global->LDS (no VGPR round-trip)
      if (t < 128) {
        const int akey = t >> 2;
        const int c8   = (t & 3) * 8;
        const _Float16* gsrc = K + ((long)b * Nn + kb + akey) * Dn + h * dh + c8;
        __builtin_amdgcn_global_load_async_to_lds_b128(
            (as1_v4i_ptr)gsrc, (as3_v4i_ptr)&kt[akey * 32 + c8], 0, 0);
      }
      wait_asynccnt0();
#else
      *reinterpret_cast<v4h*>(&kt[key * 32 + c4]) =
          *reinterpret_cast<const v4h*>(K + src);
#endif
    }
    __syncthreads();

    // ---- scores: two 16x16 tiles over this 32-key block ----
    v16h kb0 = ld_a_pair(&kt[lr * 32 + lh * 8], &kt[lr * 32 + lh * 8 + 16]);
    v16h kb1 = ld_a_pair(&kt[(16 + lr) * 32 + lh * 8], &kt[(16 + lr) * 32 + lh * 8 + 16]);
    v8f s0 = {}, s1 = {};
    s0 = wmma_f16(qa, kb0, s0);
    s1 = wmma_f16(qa, kb1, s1);

    // ---- sigmoid, then D-layout -> A-layout via per-wave LDS scratch ----
#pragma unroll
    for (int v = 0; v < 8; v++) {
      int row = v + 8 * lh;
      float a0 = 1.0f / (1.0f + __expf(-s0[v] * scale));
      float a1 = 1.0f / (1.0f + __expf(-s1[v] * scale));
      sc[wave][row * 32 + lr]      = (_Float16)a0;
      sc[wave][row * 32 + 16 + lr] = (_Float16)a1;
    }
    v16h av = ld_a_pair(&sc[wave][lr * 32 + lh * 8],
                        &sc[wave][lr * 32 + 16 + lh * 8]);

    // ---- O += A' (16x32) @ Vh (32 x 32dims) : two WMMAs from transposed V ----
    v16h vb0 = ld_a_pair(&vt[lr * 32 + lh * 8], &vt[lr * 32 + lh * 8 + 16]);
    v16h vb1 = ld_a_pair(&vt[(16 + lr) * 32 + lh * 8], &vt[(16 + lr) * 32 + lh * 8 + 16]);
    acc0 = wmma_f16(av, vb0, acc0);
    acc1 = wmma_f16(av, vb1, acc1);

    __syncthreads();
  }

  // ---- epilogue: residual Qh, store f16 ----
#pragma unroll
  for (int v = 0; v < 8; v++) {
    long row = rowbase + v + 8 * lh;
    int c0 = h * dh + lr, c1 = c0 + 16;
    float o0 = acc0[v] + (float)Q[row * Dn + c0];
    float o1 = acc1[v] + (float)Q[row * Dn + c1];
    O[row * Dn + c0] = (_Float16)o0;
    O[row * Dn + c1] = (_Float16)o1;
  }
}

// ------------------------------ PMA tail (tiny) -----------------------------
// 1 seed query per batch: Qp = S@Wq+bq; A = sigmoid(Qp Kp^T/sqrt(128));
// O = Qp + A Vp; P = O + relu(O@Wo+bo); out = P@pW+pb.
__global__ __launch_bounds__(256)
void k_pma(const float* __restrict__ S, const float* __restrict__ Wq,
           const float* __restrict__ bq, const float* __restrict__ Wo,
           const float* __restrict__ bo, const _Float16* __restrict__ Kp,
           const _Float16* __restrict__ Vp, const float* __restrict__ pW,
           const float* __restrict__ pb, float* __restrict__ outp,
           int Nn, int Dd, int Dout) {
  const int b = blockIdx.x;
  const int t = threadIdx.x;
  __shared__ float q[128], o[128], p[128];
  __shared__ float aat[4 * 2048];
  const float scale = 0.08838834764831845f;

  if (t < 128) {
    float s = bq[t];
    for (int k = 0; k < 128; k++) s += S[k] * Wq[k * 128 + t];
    q[t] = s;
  }
  __syncthreads();

  for (int m = t; m < Nn; m += blockDim.x) {
    const _Float16* krow = Kp + ((long)b * Nn + m) * Dd;
    for (int h = 0; h < 4; h++) {
      float d = 0.0f;
      for (int j = 0; j < 32; j++) d += q[h * 32 + j] * (float)krow[h * 32 + j];
      aat[h * 2048 + m] = 1.0f / (1.0f + __expf(-d * scale));
    }
  }
  __syncthreads();

  if (t < 128) {
    int h = t >> 5;
    float acc = q[t];
    for (int m = 0; m < Nn; m++)
      acc += aat[h * 2048 + m] * (float)Vp[((long)b * Nn + m) * Dd + t];
    o[t] = acc;
  }
  __syncthreads();

  if (t < 128) {
    float f = bo[t];
    for (int k = 0; k < 128; k++) f += o[k] * Wo[k * 128 + t];
    p[t] = o[t] + fmaxf(f, 0.0f);
  }
  __syncthreads();

  if (t < Dout) {
    float y = pb[t];
    for (int k = 0; k < 128; k++) y += p[k] * pW[k * Dout + t];
    outp[b * Dout + t] = y;
  }
}

// ------------------------------- launcher -----------------------------------

extern "C" void kernel_launch(void* const* d_in, const int* in_sizes, int n_in,
                              void* d_out, int out_size, void* d_ws, size_t ws_size,
                              hipStream_t stream) {
  (void)in_sizes; (void)n_in; (void)out_size; (void)ws_size;
  constexpr int B_ = 4, N_ = 2048, D_ = 128, DOUT = 256;
  constexpr int MN = B_ * N_;              // 8192 rows

  // Input order: X, {m_i: Wq,bq,Wk,bk,Wv,bv,Wo,bo}x3, S, pW, pb
  const float* X = (const float*)d_in[0];
  auto Wf = [&](int layer, int j) {        // j: 0=Wq 1=Wk 2=Wv 3=Wo
    return (const float*)d_in[1 + layer * 8 + j * 2];
  };
  auto Bf = [&](int layer, int j) {
    return (const float*)d_in[1 + layer * 8 + j * 2 + 1];
  };
  const float* S  = (const float*)d_in[25];
  const float* pW = (const float*)d_in[26];
  const float* pb = (const float*)d_in[27];

  char* ws = (char*)d_ws;
  const size_t SZ = (size_t)MN * D_ * sizeof(_Float16);   // 2 MB per activation
  _Float16* X16 = (_Float16*)(ws);
  _Float16* Q16 = (_Float16*)(ws + 1 * SZ);
  _Float16* K16 = (_Float16*)(ws + 2 * SZ);
  _Float16* V16 = (_Float16*)(ws + 3 * SZ);
  _Float16* O16 = (_Float16*)(ws + 4 * SZ);
  _Float16* H0  = (_Float16*)(ws + 5 * SZ);
  _Float16* H1  = (_Float16*)(ws + 6 * SZ);
  _Float16* WT  = (_Float16*)(ws + 7 * SZ);               // 12 x 128x128 f16 (transposed)
  auto WTp = [&](int layer, int j) { return WT + (size_t)(layer * 4 + j) * D_ * D_; };

  // Convert X and all 12 square weights (transposed for contiguous B-reads)
  {
    int n = MN * D_;
    k_cvt<<<(n + 255) / 256, 256, 0, stream>>>(X, X16, n);
    for (int l = 0; l < 3; l++)
      for (int j = 0; j < 4; j++)
        k_cvtT<<<(D_ * D_ + 255) / 256, 256, 0, stream>>>(Wf(l, j), WTp(l, j), D_, D_);
  }

  // 16x64 strip per wave, 8 waves per block
  const int gemm_blocks = (MN / 16) * (D_ / 64) / 8;      // 128
  const dim3 gblk(256);
  const dim3 ablk(256);
  const dim3 agrid(B_ * 4 * (N_ / 128));

  auto run_sab = [&](const _Float16* in16, int l, _Float16* out16) {
    k_gemm<<<gemm_blocks, gblk, 0, stream>>>(in16, WTp(l, 0), Bf(l, 0), Q16, nullptr, MN, D_, D_, 0);
    k_gemm<<<gemm_blocks, gblk, 0, stream>>>(in16, WTp(l, 1), Bf(l, 1), K16, nullptr, MN, D_, D_, 0);
    k_gemm<<<gemm_blocks, gblk, 0, stream>>>(in16, WTp(l, 2), Bf(l, 2), V16, nullptr, MN, D_, D_, 0);
    k_attn<<<agrid, ablk, 0, stream>>>(Q16, K16, V16, O16, B_, N_, D_);
    k_gemm<<<gemm_blocks, gblk, 0, stream>>>(O16, WTp(l, 3), Bf(l, 3), out16, O16, MN, D_, D_, 1);
  };

  run_sab(X16, 0, H0);      // SAB layer 0
  run_sab(H0, 1, H1);       // SAB layer 1

  // PMA: K/V projections over the set with m2 weights, then the tiny 1-seed tail.
  k_gemm<<<gemm_blocks, gblk, 0, stream>>>(H1, WTp(2, 1), Bf(2, 1), K16, nullptr, MN, D_, D_, 0);
  k_gemm<<<gemm_blocks, gblk, 0, stream>>>(H1, WTp(2, 2), Bf(2, 2), V16, nullptr, MN, D_, D_, 0);
  k_pma<<<B_, 256, 0, stream>>>(S, Wf(2, 0), Bf(2, 0), Wf(2, 3), Bf(2, 3),
                                K16, V16, pW, pb, (float*)d_out, N_, D_, DOUT);
}
